// Block_67637144977771
// MI455X (gfx1250) — compile-verified, matrix operand order checked by reading, not run on hardware
//
#include <hip/hip_runtime.h>
#include <math.h>

// ---------------------------------------------------------------------------
// CDNA5 (gfx1250) implementation of the dual-branch shifted-window attention
// block. All matmuls run on v_wmma_f32_16x16x32_bf16 (wave32 WMMA); GEMM tiles
// are staged into LDS by the Tensor Data Mover (TENSORcnt double buffering).
//
// WMMA 16-bit fragment layout (CDNA5 ISA §7.12.2):
//   A (16x32): lane l -> row M = l&15;  element e -> K = (e<8 ? e : e+8) + 8*(l>>4)
//   B (32x16): lane l -> col N = l&15;  element e -> K = (e<8 ? e : e+8) + 8*(l>>4)
//   C (16x16 f32): VGPR v, lane l -> M = v + 8*(l>>4), N = l&15
// A fragment = two contiguous 8-half (16B) runs => two b128 loads.
// ---------------------------------------------------------------------------

typedef __attribute__((ext_vector_type(16))) __bf16 v16bf;
typedef __attribute__((ext_vector_type(8)))  float  v8f;
typedef unsigned int u32x4 __attribute__((ext_vector_type(4)));
typedef int          i32x4 __attribute__((ext_vector_type(4)));
typedef int          i32x8 __attribute__((ext_vector_type(8)));

#define DIMC   512
#define CH     256
#define NHH    8
#define HD     32
#define NTOK   64
#define MROWS  32768            // B * 64 * 64
#define ATT_SCALE 0.17677669529663687f   // 32^-0.5

#if defined(__gfx1250__) && __has_builtin(__builtin_amdgcn_tensor_load_to_lds) && \
    __has_builtin(__builtin_amdgcn_s_wait_tensorcnt)
#define ATHENA_TDM 1
#endif

__device__ __forceinline__ unsigned short f2bf(float f) {
  unsigned int u = __float_as_uint(f);
  u += 0x7FFFu + ((u >> 16) & 1u);      // round-to-nearest-even
  return (unsigned short)(u >> 16);
}

__device__ __forceinline__ v16bf load_frag16(const unsigned short* p) {
  v16bf f;
  ((uint4*)&f)[0] = *(const uint4*)(p);        // K = base .. base+7
  ((uint4*)&f)[1] = *(const uint4*)(p + 16);   // K = base+16 .. base+23
  return f;
}

#ifdef ATHENA_TDM
// One TDM descriptor: copy a (tileRows x tileK) 2-byte-element tile from a
// row-major matrix (leading dim ldElems) into LDS, packed row-major
// (tileK elements per row). CDNA5 ISA.
__device__ __forceinline__ void tdm_load_tile_2d(unsigned ldsAddr,
                                                 const void* gptr, int tileK,
                                                 int tileRows, int ldElems) {
  const unsigned long long ga = (unsigned long long)gptr;
  u32x4 g0;
  g0[0] = 1u;                                        // count=1, user mode
  g0[1] = ldsAddr;                                   // lds_addr (bytes)
  g0[2] = (unsigned)ga;                              // global_addr[31:0]
  g0[3] = (unsigned)((ga >> 32) & 0x01FFFFFFull) | (2u << 30);  // [56:32]|type=2
  i32x8 g1;
  g1[0] = (1 << 16);                                 // data_size = 2 bytes
  g1[1] = (ldElems & 0xFFFF) << 16;                  // tensor_dim0[15:0]
  g1[2] = (ldElems >> 16) & 0xFFFF;                  // tensor_dim0[31:16]
  g1[3] = (1 << 4) | (tileK << 16);                  // tensor_dim1=1M, tile_dim0
  g1[4] = tileRows & 0xFFFF;                         // tile_dim1, tile_dim2=0
  g1[5] = ldElems;                                   // tensor_dim0_stride[31:0]
  g1[6] = 0;
  g1[7] = 0;
  const i32x4 z4 = {0, 0, 0, 0};
#if __clang_major__ >= 23
  const i32x8 z8 = {0, 0, 0, 0, 0, 0, 0, 0};
  __builtin_amdgcn_tensor_load_to_lds(g0, g1, z4, z4, z8, 0);
#else
  __builtin_amdgcn_tensor_load_to_lds(g0, g1, z4, z4, 0);
#endif
}
#endif

// ---------------------------------------------------------------------------
// Elementwise helpers
// ---------------------------------------------------------------------------

// out[n*K + k] = bf16(in[k*N + n])  (weights -> (N,K) bf16, K-contiguous)
__global__ void transpose_bf16(const float* __restrict__ in,
                               unsigned short* __restrict__ out,
                               int K, int N) {
  int idx = blockIdx.x * 256 + threadIdx.x;
  if (idx >= K * N) return;
  int k = idx / N, n = idx - k * N;
  out[(size_t)n * K + k] = f2bf(in[idx]);
}

// row-wise pixel norm over 512 channels, emit bf16
__global__ __launch_bounds__(256, 1)
void pixelnorm_bf16(const float* __restrict__ x,
                    unsigned short* __restrict__ out) {
  const int row = blockIdx.x;
  const float* p = x + (size_t)row * DIMC;
  float a0 = p[threadIdx.x];
  float a1 = p[threadIdx.x + 256];
  float ss = a0 * a0 + a1 * a1;
#pragma unroll
  for (int m = 1; m < 32; m <<= 1) ss += __shfl_xor(ss, m, 32);
  __shared__ float part[8];
  if ((threadIdx.x & 31) == 0) part[threadIdx.x >> 5] = ss;
  __syncthreads();
  float tot = part[0] + part[1] + part[2] + part[3] +
              part[4] + part[5] + part[6] + part[7];
  float r = rsqrtf(tot * (1.0f / DIMC) + 1e-8f);
  unsigned short* o = out + (size_t)row * DIMC;
  o[threadIdx.x]       = f2bf(a0 * r);
  o[threadIdx.x + 256] = f2bf(a1 * r);
}

// ---------------------------------------------------------------------------
// bf16 WMMA GEMM with TDM-staged, double-buffered LDS tiles.
//   C[M,N] = A[M,K] @ Bt[N,K]^T (+epilogue)
// block = 256 threads = 8 waves (2 row-waves x 4 col-waves)
// wave tile 64x64 -> block tile 128x256; grid = (N/256, M/128)
// Per K-step: wave0 TDM-loads B tile (256x32), wave1 A tile (128x32) into the
// ping-pong buffer for step i+1, waits TENSORcnt<=1 (step-i tile done),
// barrier, then all waves run 16 WMMAs from LDS fragments.
// EPI 0: bias->bf16 ; EPI 1: bias+residual->f32 ; EPI 2: bias+GELU->bf16
// ---------------------------------------------------------------------------
template <int EPI>
__global__ __launch_bounds__(256, 1)
void gemm_bf16(const unsigned short* __restrict__ A,
               const unsigned short* __restrict__ Bt,
               const float* __restrict__ bias,
               const float* __restrict__ res,
               unsigned short* __restrict__ outb,
               float* __restrict__ outf,
               int N, int K) {
  __shared__ unsigned short Ast[2][128 * 32];   // 2 x 8 KB
  __shared__ unsigned short Bst[2][256 * 32];   // 2 x 16 KB

  const int tid  = threadIdx.x;
  const int l    = tid & 31;
  const int w    = tid >> 5;
  const int lo16 = l & 15;
  const int hi   = l >> 4;
  const int mBlock = blockIdx.y * 128;
  const int nBlock = blockIdx.x * 256;
  const int mBase  = mBlock + (w >> 2) * 64;
  const int nBase  = nBlock + (w & 3) * 64;

  auto stage = [&](int buf, int kb) {
#ifdef ATHENA_TDM
    if (w == 0)
      tdm_load_tile_2d((unsigned)(unsigned long long)&Bst[buf][0],
                       Bt + (size_t)nBlock * K + kb, 32, 256, K);
    else if (w == 1)
      tdm_load_tile_2d((unsigned)(unsigned long long)&Ast[buf][0],
                       A + (size_t)mBlock * K + kb, 32, 128, K);
#else
    for (int j = tid; j < 1024; j += 256) {  // B: 256 rows x 4 uint4
      const int n = j >> 2, c = j & 3;
      ((uint4*)&Bst[buf][0])[j] =
          *(const uint4*)(Bt + (size_t)(nBlock + n) * K + kb + c * 8);
    }
    for (int j = tid; j < 512; j += 256) {   // A: 128 rows x 4 uint4
      const int n = j >> 2, c = j & 3;
      ((uint4*)&Ast[buf][0])[j] =
          *(const uint4*)(A + (size_t)(mBlock + n) * K + kb + c * 8);
    }
#endif
  };

  const v8f vzero = {0.f, 0.f, 0.f, 0.f, 0.f, 0.f, 0.f, 0.f};
  v8f acc[4][4];
#pragma unroll
  for (int i = 0; i < 4; i++)
#pragma unroll
    for (int j = 0; j < 4; j++) acc[i][j] = vzero;

  stage(0, 0);
  const int nk = K >> 5;
  for (int i = 0; i < nk; i++) {
    const int cur = i & 1;
    if (i + 1 < nk) {
      stage(1 - cur, (i + 1) << 5);
#ifdef ATHENA_TDM
      __builtin_amdgcn_s_wait_tensorcnt((short)1);  // oldest (step i) done
#endif
    } else {
#ifdef ATHENA_TDM
      __builtin_amdgcn_s_wait_tensorcnt((short)0);
#endif
    }
    __syncthreads();

    v16bf af[4], bf[4];
#pragma unroll
    for (int tm = 0; tm < 4; tm++)
      af[tm] = load_frag16(
          &Ast[cur][((w >> 2) * 64 + tm * 16 + lo16) * 32 + 8 * hi]);
#pragma unroll
    for (int tn = 0; tn < 4; tn++)
      bf[tn] = load_frag16(
          &Bst[cur][((w & 3) * 64 + tn * 16 + lo16) * 32 + 8 * hi]);
#pragma unroll
    for (int tm = 0; tm < 4; tm++)
#pragma unroll
      for (int tn = 0; tn < 4; tn++)
        acc[tm][tn] = __builtin_amdgcn_wmma_f32_16x16x32_bf16(
            false, af[tm], false, bf[tn], (short)0, acc[tm][tn], false, false);

    __syncthreads();  // all reads of buffer `cur` complete before re-staging
  }

#pragma unroll
  for (int tn = 0; tn < 4; tn++) {
    const int n = nBase + tn * 16 + lo16;
    const float bv = bias[n];
#pragma unroll
    for (int tm = 0; tm < 4; tm++) {
#pragma unroll
      for (int v = 0; v < 8; v++) {
        const int m = mBase + tm * 16 + v + 8 * hi;
        const size_t idx = (size_t)m * N + n;
        const float a = acc[tm][tn][v] + bv;
        if (EPI == 0) {
          outb[idx] = f2bf(a);
        } else if (EPI == 1) {
          outf[idx] = a + res[idx];
        } else {  // exact GELU
          const float g = 0.5f * a * (1.0f + erff(a * 0.70710678118654752f));
          outb[idx] = f2bf(g);
        }
      }
    }
  }
}

// ---------------------------------------------------------------------------
// Windowed attention, both branches. grid.x = 1024: [branch(2)][batch(8)][win(64)]
// block = 256 threads = 8 waves; wave == head. Branch 1 gathers channels
// [256,512) with the (+4,+4) roll folded into index math; scatter through the
// same coordinates realizes window_reverse + inverse roll.
// ---------------------------------------------------------------------------
__global__ __launch_bounds__(256, 1)
void win_attn(const unsigned short* __restrict__ qkv,   // (32768, 1536) bf16
              const float* __restrict__ rpb1,           // (225, 8)
              const float* __restrict__ rpb2,
              unsigned short* __restrict__ outp) {      // (32768, 512) bf16
  __shared__ unsigned short attn_s[NHH][NTOK][NTOK];    // 64 KB

  const int bid  = blockIdx.x;
  const int br   = bid >> 9;
  const int widx = bid & 511;
  const int b    = widx >> 6;
  const int wid  = widx & 63;
  const int wy   = wid >> 3, wx = wid & 7;
  const int sh   = br ? 4 : 0;

  const int l    = threadIdx.x & 31;
  const int h    = threadIdx.x >> 5;
  const int lo16 = l & 15, hi = l >> 4;
  const int d0   = br * CH + h * HD;
  const float* rpb = br ? rpb2 : rpb1;

  auto trow = [&](int tok) -> int {   // window token -> global qkv row
    const int gy = ((wy << 3) + (tok >> 3) + sh) & 63;
    const int gx = ((wx << 3) + (tok & 7) + sh) & 63;
    return (b << 12) + (gy << 6) + gx;
  };
  auto lab = [](int g) -> int { return g < 56 ? 0 : (g < 60 ? 1 : 2); };

  const v8f vzero = {0.f, 0.f, 0.f, 0.f, 0.f, 0.f, 0.f, 0.f};

  // ---- scores = Q @ K^T  (M=64 tokens, N=64 tokens, K=32 ch) ----
  v16bf qf[4], kf[4];
#pragma unroll
  for (int tm = 0; tm < 4; tm++) {
    const int r = trow(tm * 16 + lo16);
    qf[tm] = load_frag16(qkv + (size_t)r * 1536 + 0 * DIMC + d0 + 8 * hi);
  }
#pragma unroll
  for (int tn = 0; tn < 4; tn++) {
    const int r = trow(tn * 16 + lo16);
    kf[tn] = load_frag16(qkv + (size_t)r * 1536 + 1 * DIMC + d0 + 8 * hi);
  }
  v8f acc[4][4];
#pragma unroll
  for (int i = 0; i < 4; i++)
#pragma unroll
    for (int j = 0; j < 4; j++) acc[i][j] = vzero;
#pragma unroll
  for (int tm = 0; tm < 4; tm++)
#pragma unroll
    for (int tn = 0; tn < 4; tn++)
      acc[tm][tn] = __builtin_amdgcn_wmma_f32_16x16x32_bf16(
          false, qf[tm], false, kf[tn], (short)0, acc[tm][tn], false, false);

  // ---- scale + rel-pos bias + shift mask + row softmax -> LDS (bf16) ----
#pragma unroll
  for (int tm = 0; tm < 4; tm++) {
#pragma unroll
    for (int v = 0; v < 8; v++) {
      const int m  = tm * 16 + v + 8 * hi;
      const int my = m >> 3, mxx = m & 7;
      const int Li = 3 * lab((wy << 3) + my) + lab((wx << 3) + mxx);
      float rowv[4];
#pragma unroll
      for (int tn = 0; tn < 4; tn++) {
        const int n  = tn * 16 + lo16;
        const int ny = n >> 3, nxx = n & 7;
        const int ridx = (my - ny) + (mxx - nxx) + 29;   // dy+7+15 + dx+7
        float a = acc[tm][tn][v] * ATT_SCALE + rpb[ridx * NHH + h];
        if (br) {
          const int Lj = 3 * lab((wy << 3) + ny) + lab((wx << 3) + nxx);
          if (Li != Lj) a -= 100.0f;
        }
        rowv[tn] = a;
      }
      float mx = fmaxf(fmaxf(rowv[0], rowv[1]), fmaxf(rowv[2], rowv[3]));
#pragma unroll
      for (int s = 1; s < 16; s <<= 1) mx = fmaxf(mx, __shfl_xor(mx, s, 32));
      float sum = 0.f;
#pragma unroll
      for (int tn = 0; tn < 4; tn++) {
        rowv[tn] = __expf(rowv[tn] - mx);
        sum += rowv[tn];
      }
#pragma unroll
      for (int s = 1; s < 16; s <<= 1) sum += __shfl_xor(sum, s, 32);
      const float inv = 1.0f / sum;
#pragma unroll
      for (int tn = 0; tn < 4; tn++)
        attn_s[h][m][tn * 16 + lo16] = f2bf(rowv[tn] * inv);
    }
  }
  // attn_s region is produced and consumed by the same wave; same-wave DS ops
  // stay in order, so no workgroup barrier is needed.

  // ---- out = P @ V  (M=64 tokens, N=32 ch, K=64 tokens) ----
  v8f o[4][2];
#pragma unroll
  for (int i = 0; i < 4; i++) { o[i][0] = vzero; o[i][1] = vzero; }
#pragma unroll
  for (int ks = 0; ks < 2; ks++) {
    v16bf bv[2];
#pragma unroll
    for (int tn = 0; tn < 2; tn++) {
      union { v16bf v; unsigned short u[16]; } t;
#pragma unroll
      for (int e = 0; e < 16; e++) {
        const int k = ks * 32 + ((e < 8) ? (e + 8 * hi) : (16 + (e - 8) + 8 * hi));
        const int r = trow(k);
        t.u[e] = qkv[(size_t)r * 1536 + 2 * DIMC + d0 + tn * 16 + lo16];
      }
      bv[tn] = t.v;
    }
#pragma unroll
    for (int tm = 0; tm < 4; tm++) {
      const v16bf af =
          load_frag16(&attn_s[h][tm * 16 + lo16][ks * 32 + 8 * hi]);
#pragma unroll
      for (int tn = 0; tn < 2; tn++)
        o[tm][tn] = __builtin_amdgcn_wmma_f32_16x16x32_bf16(
            false, af, false, bv[tn], (short)0, o[tm][tn], false, false);
    }
  }

  // ---- scatter (window_reverse + inverse roll == same coords) ----
#pragma unroll
  for (int tm = 0; tm < 4; tm++) {
#pragma unroll
    for (int tn = 0; tn < 2; tn++) {
#pragma unroll
      for (int v = 0; v < 8; v++) {
        const int m = tm * 16 + v + 8 * hi;
        const int r = trow(m);
        const int col = br * CH + h * HD + tn * 16 + lo16;
        outp[(size_t)r * DIMC + col] = f2bf(o[tm][tn][v]);
      }
    }
  }
}

// ---------------------------------------------------------------------------
// Host-side orchestration
// d_in: 0:x 1:qkv_w 2:qkv_b 3:proj_w 4:proj_b 5:rpb1 6:rpb2
//       7:fc1_w 8:fc1_b 9:fc2_w 10:fc2_b
// ---------------------------------------------------------------------------
extern "C" void kernel_launch(void* const* d_in, const int* in_sizes, int n_in,
                              void* d_out, int out_size, void* d_ws,
                              size_t ws_size, hipStream_t stream) {
  const float* x      = (const float*)d_in[0];
  const float* qkv_w  = (const float*)d_in[1];
  const float* qkv_b  = (const float*)d_in[2];
  const float* proj_w = (const float*)d_in[3];
  const float* proj_b = (const float*)d_in[4];
  const float* rpb1   = (const float*)d_in[5];
  const float* rpb2   = (const float*)d_in[6];
  const float* fc1_w  = (const float*)d_in[7];
  const float* fc1_b  = (const float*)d_in[8];
  const float* fc2_w  = (const float*)d_in[9];
  const float* fc2_b  = (const float*)d_in[10];

  char* ws = (char*)d_ws;
  // weight transposes (bf16, (N,K))
  unsigned short* qkv_wT = (unsigned short*)(ws + 0ull);          // 1.5 MB
  unsigned short* proj_wT= (unsigned short*)(ws + 1572864ull);    // 0.5 MB
  unsigned short* fc1_wT = (unsigned short*)(ws + 2097152ull);    // 2 MB
  unsigned short* fc2_wT = (unsigned short*)(ws + 4194304ull);    // 2 MB
  // activations (lifetimes overlapped)
  unsigned short* xn   = (unsigned short*)(ws + 6291456ull);      // 32 MB (xn / yn)
  unsigned short* qkvb = (unsigned short*)(ws + 39845888ull);     // 128 MB (qkv / mlp hidden)
  unsigned short* attno= (unsigned short*)(ws + 174063616ull);    // 32 MB
  float*          xo   = (float*)(ws + 207618048ull);             // 64 MB
  float*          out  = (float*)d_out;

  // 1) weights -> (N,K) bf16
  transpose_bf16<<<(512 * 1536 + 255) / 256, 256, 0, stream>>>(qkv_w, qkv_wT, 512, 1536);
  transpose_bf16<<<(512 * 512 + 255) / 256, 256, 0, stream>>>(proj_w, proj_wT, 512, 512);
  transpose_bf16<<<(512 * 2048 + 255) / 256, 256, 0, stream>>>(fc1_w, fc1_wT, 512, 2048);
  transpose_bf16<<<(2048 * 512 + 255) / 256, 256, 0, stream>>>(fc2_w, fc2_wT, 2048, 512);

  // 2) pixel_norm(x) -> bf16
  pixelnorm_bf16<<<MROWS, 256, 0, stream>>>(x, xn);

  // 3) qkv = xn @ qkv_w + b   (32768 x 1536, K=512) -> bf16
  gemm_bf16<0><<<dim3(1536 / 256, MROWS / 128), 256, 0, stream>>>(
      xn, qkv_wT, qkv_b, nullptr, qkvb, nullptr, 1536, 512);

  // 4) dual-branch window attention -> attno (32768 x 512) bf16
  win_attn<<<1024, 256, 0, stream>>>(qkvb, rpb1, rpb2, attno);

  // 5) xo = attno @ proj_w + b + x   (f32 residual kept)
  gemm_bf16<1><<<dim3(512 / 256, MROWS / 128), 256, 0, stream>>>(
      attno, proj_wT, proj_b, x, nullptr, xo, 512, 512);

  // 6) yn = pixel_norm(xo) -> bf16 (reuses xn region)
  pixelnorm_bf16<<<MROWS, 256, 0, stream>>>(xo, xn);

  // 7) h = gelu(yn @ fc1_w + b)  (32768 x 2048) -> bf16 (reuses qkv region)
  gemm_bf16<2><<<dim3(2048 / 256, MROWS / 128), 256, 0, stream>>>(
      xn, fc1_wT, fc1_b, nullptr, qkvb, nullptr, 2048, 512);

  // 8) out = xo + h @ fc2_w + b  (32768 x 512, K=2048) -> f32
  gemm_bf16<1><<<dim3(512 / 256, MROWS / 128), 256, 0, stream>>>(
      qkvb, fc2_wT, fc2_b, xo, nullptr, out, 512, 2048);
}
